// CrossAttnBlock_47098611368061
// MI455X (gfx1250) — matrix operand order repaired and checked by reading, not compiled, not run on hardware
//
#include <hip/hip_runtime.h>

// ---------------------------------------------------------------------------
// CDNA5 (gfx1250) transformer block: cross-attn -> self-attn -> MLP.
// bf16 WMMA (v_wmma_f32_16x16x32_bf16) everywhere, fp32 accumulate.
// Flash-attention streaming softmax (never materializes [b,h,n,n] scores).
// GEMM: BK=64, register double-buffered mainloop, templated epilogue.
// ---------------------------------------------------------------------------

typedef __attribute__((ext_vector_type(16))) __bf16 v16bf;
typedef __attribute__((ext_vector_type(8)))  __bf16 v8bf;
typedef __attribute__((ext_vector_type(2)))  __bf16 v2bf;
typedef __attribute__((ext_vector_type(8)))  float  v8f;

#define WMMA_BF16(A, B, C) \
    __builtin_amdgcn_wmma_f32_16x16x32_bf16(false, (A), false, (B), (short)0, (C), false, false)

static __device__ __forceinline__ v16bf cat8(v8bf lo, v8bf hi) {
    return __builtin_shufflevector(lo, hi, 0,1,2,3,4,5,6,7,8,9,10,11,12,13,14,15);
}

static __device__ __forceinline__ float gelu_f(float x) {
    float x3 = x * x * x;
    return 0.5f * x * (1.0f + tanhf(0.7978845608028654f * (x + 0.044715f * x3)));
}

// ---------------------------------------------------------------------------
// LayerNorm over rows of length 1024, fp32 in -> bf16 out. One block per row.
// ---------------------------------------------------------------------------
__global__ __launch_bounds__(256) void ln_kernel(const float* __restrict__ x,
                                                 const float* __restrict__ sc,
                                                 const float* __restrict__ bi,
                                                 __bf16* __restrict__ out) {
    const int row = blockIdx.x;
    const int tid = threadIdx.x;
    const float* xr = x + (size_t)row * 1024;
    float4 v = *(const float4*)(xr + tid * 4);
    float s  = v.x + v.y + v.z + v.w;
    float s2 = v.x * v.x + v.y * v.y + v.z * v.z + v.w * v.w;
#pragma unroll
    for (int m = 16; m >= 1; m >>= 1) {
        s  += __shfl_xor(s, m);
        s2 += __shfl_xor(s2, m);
    }
    __shared__ float red[16];
    const int wv = tid >> 5;
    if ((tid & 31) == 0) { red[wv] = s; red[8 + wv] = s2; }
    __syncthreads();
    float tot = 0.f, tot2 = 0.f;
#pragma unroll
    for (int i = 0; i < 8; ++i) { tot += red[i]; tot2 += red[8 + i]; }
    const float mean = tot * (1.0f / 1024.0f);
    const float var  = tot2 * (1.0f / 1024.0f) - mean * mean;
    const float inv  = rsqrtf(var + 1e-6f);
    float4 sv = *(const float4*)(sc + tid * 4);
    float4 bv = *(const float4*)(bi + tid * 4);
    __bf16* o = out + (size_t)row * 1024 + tid * 4;
    o[0] = (__bf16)((v.x - mean) * inv * sv.x + bv.x);
    o[1] = (__bf16)((v.y - mean) * inv * sv.y + bv.y);
    o[2] = (__bf16)((v.z - mean) * inv * sv.z + bv.z);
    o[3] = (__bf16)((v.w - mean) * inv * sv.w + bv.w);
}

// ---------------------------------------------------------------------------
// Transpose + fp32->bf16 convert:  W[K][N] (row major) -> Wt[N][K] (row major)
// ---------------------------------------------------------------------------
__global__ __launch_bounds__(256) void transpose_cvt_kernel(const float* __restrict__ W,
                                                            __bf16* __restrict__ Wt,
                                                            int K, int N) {
    __shared__ float tile[32][33];
    const int nb = blockIdx.x * 32, kb = blockIdx.y * 32;
    const int tx = threadIdx.x, ty = threadIdx.y;
#pragma unroll
    for (int i = 0; i < 4; ++i)
        tile[ty + i * 8][tx] = W[(size_t)(kb + ty + i * 8) * N + nb + tx];
    __syncthreads();
#pragma unroll
    for (int i = 0; i < 4; ++i)
        Wt[(size_t)(nb + ty + i * 8) * K + kb + tx] = (__bf16)tile[tx][ty + i * 8];
}

// ---------------------------------------------------------------------------
// GEMM: out[M,N] = A[M,K](bf16,row) @ Bt[N,K](bf16,row, i.e. W transposed)
// 128x128x64 tiles, 256 threads = 8 waves; wave computes 32x64 output
// (2x4 WMMA tiles, 16 WMMAs per staged tile = 2 k-steps per barrier pair).
// Register double-buffer: prefetch global tile k+1 while WMMAs consume tile k.
// MODE: 0 = f32 store   1 = bf16 store   2 = gelu -> bf16
// ---------------------------------------------------------------------------
template <int MODE, bool HAS_BIAS, bool HAS_RESID>
__global__ __launch_bounds__(256) void gemm_bf16_kernel(const __bf16* __restrict__ A,
                                                        const __bf16* __restrict__ Bt,
                                                        const float* __restrict__ bias,
                                                        const float* __restrict__ resid,
                                                        void* __restrict__ outp,
                                                        int N, int K) {
    __shared__ __align__(16) __bf16 As[128 * 72];
    __shared__ __align__(16) __bf16 Bs[128 * 72];
    const int tid  = threadIdx.x;
    const int wv   = tid >> 5;
    const int lane = tid & 31;
    const int lm   = lane & 15;
    const int lh   = lane >> 4;
    const int wRow = wv >> 1;   // 0..3  (rows of 32)
    const int wCol = wv & 1;    // 0..1  (cols of 64)
    const int mBase = blockIdx.y * 128;
    const int nBase = blockIdx.x * 128;

    // Per-thread staging: 4 chunks of A + 4 of B (128 rows x 64 cols bf16).
    const __bf16* Ag[4];
    const __bf16* Bg[4];
    int ldsOff[4];
#pragma unroll
    for (int i = 0; i < 4; ++i) {
        const int l   = tid + i * 256;
        const int row = l >> 3;          // 0..127
        const int col = (l & 7) << 3;    // 0,8,...,56
        Ag[i] = A  + (size_t)(mBase + row) * K + col;
        Bg[i] = Bt + (size_t)(nBase + row) * K + col;
        ldsOff[i] = row * 72 + col;
    }

    v8f acc[2][4];
#pragma unroll
    for (int mt = 0; mt < 2; ++mt)
#pragma unroll
        for (int nt = 0; nt < 4; ++nt)
#pragma unroll
            for (int r = 0; r < 8; ++r) acc[mt][nt][r] = 0.0f;

    // Prefetch tile 0.
    v8bf ra[4], rb[4];
#pragma unroll
    for (int i = 0; i < 4; ++i) {
        ra[i] = *(const v8bf*)(Ag[i]);
        rb[i] = *(const v8bf*)(Bg[i]);
    }

    for (int k0 = 0; k0 < K; k0 += 64) {
        // Commit current tile to LDS.
#pragma unroll
        for (int i = 0; i < 4; ++i) {
            *(v8bf*)&As[ldsOff[i]] = ra[i];
            *(v8bf*)&Bs[ldsOff[i]] = rb[i];
        }
        __syncthreads();

        // Prefetch next tile from global (overlaps with WMMAs below).
        if (k0 + 64 < K) {
#pragma unroll
            for (int i = 0; i < 4; ++i) {
                ra[i] = *(const v8bf*)(Ag[i] + k0 + 64);
                rb[i] = *(const v8bf*)(Bg[i] + k0 + 64);
            }
        }

#pragma unroll
        for (int s = 0; s < 2; ++s) {
            v16bf af[2], bfrg[4];
#pragma unroll
            for (int mt = 0; mt < 2; ++mt) {
                const int ar = (wRow * 32 + mt * 16 + lm) * 72 + s * 32;
                af[mt] = cat8(*(const v8bf*)&As[ar + lh * 8],
                              *(const v8bf*)&As[ar + 16 + lh * 8]);
            }
#pragma unroll
            for (int nt = 0; nt < 4; ++nt) {
                const int br = (wCol * 64 + nt * 16 + lm) * 72 + s * 32;
                bfrg[nt] = cat8(*(const v8bf*)&Bs[br + lh * 8],
                                *(const v8bf*)&Bs[br + 16 + lh * 8]);
            }
#pragma unroll
            for (int mt = 0; mt < 2; ++mt)
#pragma unroll
                for (int nt = 0; nt < 4; ++nt)
                    acc[mt][nt] = WMMA_BF16(af[mt], bfrg[nt], acc[mt][nt]);
        }

        __syncthreads();  // all LDS reads done before next iteration's stores
    }

    // Branch-free epilogue. C layout: VGPR r, lane L -> row = r + 8*(L>>4), col = L&15.
    float* __restrict__ outF = (float*)outp;
    __bf16* __restrict__ outB = (__bf16*)outp;
#pragma unroll
    for (int mt = 0; mt < 2; ++mt)
#pragma unroll
        for (int nt = 0; nt < 4; ++nt) {
            const int row0 = mBase + wRow * 32 + mt * 16 + lh * 8;
            const int col  = nBase + wCol * 64 + nt * 16 + lm;
            float bc = 0.0f;
            if (HAS_BIAS) bc = bias[col];
#pragma unroll
            for (int r = 0; r < 8; ++r) {
                const size_t idx = (size_t)(row0 + r) * N + col;
                float vv = acc[mt][nt][r] + bc;
                if (HAS_RESID) vv += resid[idx];
                if (MODE == 0) {
                    outF[idx] = vv;
                } else if (MODE == 1) {
                    outB[idx] = (__bf16)vv;
                } else {
                    outB[idx] = (__bf16)gelu_f(vv);
                }
            }
        }
}

// ---------------------------------------------------------------------------
// Flash attention, head dim 64, NK = 1024, softmax scale 1/8.
// Block = 256 threads (8 waves) handles 128 queries of one (b,h).
// Streams K/V in 64-key tiles through LDS with register double-buffering.
// grid: (NQ/128, B*H)
// ---------------------------------------------------------------------------
__global__ __launch_bounds__(256) void flash_attn_kernel(const __bf16* __restrict__ Qp,
                                                         const __bf16* __restrict__ Kp,
                                                         const __bf16* __restrict__ Vp,
                                                         __bf16* __restrict__ Op,
                                                         int qStride, int kvStride) {
    __shared__ __align__(16) __bf16 Ks[64 * 72];   // [key][d]
    __shared__ __align__(16) __bf16 Vt[64 * 72];   // [d][key] (transposed)
    __shared__ __align__(16) __bf16 Ps[8 * 1024];  // per-wave 16x64 P tile
    const int tid  = threadIdx.x;
    const int wv   = tid >> 5;
    const int lane = tid & 31;
    const int lm   = lane & 15;
    const int lh   = lane >> 4;
    const int bb   = blockIdx.y >> 4;
    const int hh   = blockIdx.y & 15;

    const __bf16* q = Qp + (size_t)bb * 1024 * qStride  + hh * 64;
    const __bf16* k = Kp + (size_t)bb * 1024 * kvStride + hh * 64;
    const __bf16* v = Vp + (size_t)bb * 1024 * kvStride + hh * 64;
    __bf16*       o = Op + (size_t)bb * 1024 * 1024     + hh * 64;

    // Q fragment (A-matrix 16x64 over two 32-K steps), kept in registers.
    const int qrow = blockIdx.x * 128 + wv * 16 + lm;
    v16bf aQ[2];
#pragma unroll
    for (int s = 0; s < 2; ++s) {
        const __bf16* qr = q + (size_t)qrow * qStride + s * 32 + lh * 8;
        aQ[s] = cat8(*(const v8bf*)qr, *(const v8bf*)(qr + 16));
    }

    float mrow[8], lrow[8];
    v8f oacc[4];
#pragma unroll
    for (int r = 0; r < 8; ++r) { mrow[r] = -1e30f; lrow[r] = 0.0f; }
#pragma unroll
    for (int nt = 0; nt < 4; ++nt)
#pragma unroll
        for (int r = 0; r < 8; ++r) oacc[nt][r] = 0.0f;

    __bf16* Pw = Ps + wv * 1024;

    // K/V tile staging: thread covers K rows {kr, kr+32} and V rows {2kr, 2kr+1}.
    const int kr  = tid >> 3;        // 0..31
    const int c8  = (tid & 7) * 8;   // 0..56
    v8bf rk0, rk1, rv0, rv1;
    auto loadKV = [&](int kt) {
        const size_t base = (size_t)(kt * 64) * kvStride + c8;
        rk0 = *(const v8bf*)&k[base + (size_t)kr * kvStride];
        rk1 = *(const v8bf*)&k[base + (size_t)(kr + 32) * kvStride];
        rv0 = *(const v8bf*)&v[base + (size_t)(2 * kr) * kvStride];
        rv1 = *(const v8bf*)&v[base + (size_t)(2 * kr + 1) * kvStride];
    };
    loadKV(0);

    for (int kt = 0; kt < 16; ++kt) {
        __syncthreads();  // previous tile's LDS reads complete
        // Commit staged K (row major) and V (transposed, packed b32 stores).
        *(v8bf*)&Ks[kr * 72 + c8]        = rk0;
        *(v8bf*)&Ks[(kr + 32) * 72 + c8] = rk1;
#pragma unroll
        for (int e = 0; e < 8; ++e) {
            v2bf p;
            p[0] = rv0[e];
            p[1] = rv1[e];
            *(v2bf*)&Vt[(c8 + e) * 72 + 2 * kr] = p;
        }
        __syncthreads();

        // Prefetch next tile (overlaps with WMMAs + softmax below).
        if (kt + 1 < 16) loadKV(kt + 1);

        // S = Q @ K^T  (16 queries x 64 keys per wave)
        v8f sacc[4];
#pragma unroll
        for (int nt = 0; nt < 4; ++nt)
#pragma unroll
            for (int r = 0; r < 8; ++r) sacc[nt][r] = 0.0f;
#pragma unroll
        for (int nt = 0; nt < 4; ++nt) {
            const int base = (nt * 16 + lm) * 72;
#pragma unroll
            for (int s = 0; s < 2; ++s) {
                v16bf bK = cat8(*(const v8bf*)&Ks[base + s * 32 + lh * 8],
                                *(const v8bf*)&Ks[base + s * 32 + 16 + lh * 8]);
                sacc[nt] = WMMA_BF16(aQ[s], bK, sacc[nt]);
            }
        }

        // Online softmax per query row (row = r + 8*lh; 16 cols/lane-half).
#pragma unroll
        for (int r = 0; r < 8; ++r) {
            float mx = -1e30f;
#pragma unroll
            for (int nt = 0; nt < 4; ++nt) {
                sacc[nt][r] *= 0.125f;
                mx = fmaxf(mx, sacc[nt][r]);
            }
            mx = fmaxf(mx, __shfl_xor(mx, 1));
            mx = fmaxf(mx, __shfl_xor(mx, 2));
            mx = fmaxf(mx, __shfl_xor(mx, 4));
            mx = fmaxf(mx, __shfl_xor(mx, 8));
            const float mnew  = fmaxf(mrow[r], mx);
            const float alpha = __expf(mrow[r] - mnew);
            mrow[r] = mnew;
            float sum = 0.0f;
#pragma unroll
            for (int nt = 0; nt < 4; ++nt) {
                float p = __expf(sacc[nt][r] - mnew);
                sacc[nt][r] = p;
                sum += p;
            }
            sum += __shfl_xor(sum, 1);
            sum += __shfl_xor(sum, 2);
            sum += __shfl_xor(sum, 4);
            sum += __shfl_xor(sum, 8);
            lrow[r] = lrow[r] * alpha + sum;
#pragma unroll
            for (int nt = 0; nt < 4; ++nt) oacc[nt][r] *= alpha;
            // Stage P into wave-private LDS (C-layout -> row major 16x64).
            const int prow = (r + 8 * lh) * 64;
#pragma unroll
            for (int nt = 0; nt < 4; ++nt)
                Pw[prow + nt * 16 + lm] = (__bf16)sacc[nt][r];
        }

        // O += P @ V  (A = P 16x64, B = V 64x64 read from transposed LDS).
        v16bf aP[2];
#pragma unroll
        for (int s = 0; s < 2; ++s)
            aP[s] = cat8(*(const v8bf*)&Pw[lm * 64 + s * 32 + lh * 8],
                         *(const v8bf*)&Pw[lm * 64 + s * 32 + 16 + lh * 8]);
#pragma unroll
        for (int nt = 0; nt < 4; ++nt) {
            const int base = (nt * 16 + lm) * 72;
#pragma unroll
            for (int s = 0; s < 2; ++s) {
                v16bf bV = cat8(*(const v8bf*)&Vt[base + s * 32 + lh * 8],
                                *(const v8bf*)&Vt[base + s * 32 + 16 + lh * 8]);
                oacc[nt] = WMMA_BF16(aP[s], bV, oacc[nt]);
            }
        }
    }

    // Finalize: O / l, merged-head bf16 store.
#pragma unroll
    for (int nt = 0; nt < 4; ++nt)
#pragma unroll
        for (int r = 0; r < 8; ++r) {
            const int row = blockIdx.x * 128 + wv * 16 + r + 8 * lh;
            o[(size_t)row * 1024 + nt * 16 + lm] = (__bf16)(oacc[nt][r] / lrow[r]);
        }
}

// ---------------------------------------------------------------------------
// Host-side orchestration (stream-ordered; graph-capture safe).
// ---------------------------------------------------------------------------
extern "C" void kernel_launch(void* const* d_in, const int* in_sizes, int n_in,
                              void* d_out, int out_size, void* d_ws, size_t ws_size,
                              hipStream_t stream) {
    (void)in_sizes; (void)n_in; (void)out_size; (void)ws_size;
    const float* x     = (const float*)d_in[0];
    const float* keyv  = (const float*)d_in[1];
    const float* ln1s  = (const float*)d_in[2];
    const float* ln1b  = (const float*)d_in[3];
    const float* ln2s  = (const float*)d_in[4];
    const float* ln2b  = (const float*)d_in[5];
    const float* ln3s  = (const float*)d_in[6];
    const float* ln3b  = (const float*)d_in[7];
    const float* ln4s  = (const float*)d_in[8];
    const float* ln4b  = (const float*)d_in[9];
    const float* Wkv   = (const float*)d_in[10];
    const float* Wq    = (const float*)d_in[11];
    const float* Wco   = (const float*)d_in[12];
    const float* bco   = (const float*)d_in[13];
    const float* Wqkv  = (const float*)d_in[14];
    const float* Wso   = (const float*)d_in[15];
    const float* bso   = (const float*)d_in[16];
    const float* W1    = (const float*)d_in[17];
    const float* b1    = (const float*)d_in[18];
    const float* W2    = (const float*)d_in[19];
    const float* b2    = (const float*)d_in[20];
    float* out = (float*)d_out;

    char* ws = (char*)d_ws;
    const size_t MB = 1024ull * 1024ull;
    const size_t EL = 1024ull * 1024ull;                 // 1M elements
    __bf16* wT  = (__bf16*)(ws);                         // 32 MB bf16 weights^T
    __bf16* lnB = (__bf16*)(ws + 32 * MB);               // 8 MB  (LN out / attn out)
    __bf16* qB  = (__bf16*)(ws + 40 * MB);               // 8 MB  cross-attn Q
    __bf16* kvB = (__bf16*)(ws + 48 * MB);               // 24 MB kv / qkv
    __bf16* a2B = (__bf16*)(ws + 72 * MB);               // 8 MB  LN2 out
    float*  xrB = (float*)(ws + 80 * MB);                // 16 MB residual 1
    float*  x2B = (float*)(ws + 96 * MB);                // 16 MB residual 2
    __bf16* hB  = (__bf16*)(ws + 40 * MB);               // 32 MB MLP hidden (reuses q/kv)

    __bf16* wT_kv  = wT;
    __bf16* wT_q   = wT + 2 * EL;
    __bf16* wT_co  = wT + 3 * EL;
    __bf16* wT_qkv = wT + 4 * EL;
    __bf16* wT_so  = wT + 7 * EL;
    __bf16* wT_w1  = wT + 8 * EL;
    __bf16* wT_w2  = wT + 12 * EL;

    const dim3 tb(32, 8);
    transpose_cvt_kernel<<<dim3(64, 32),  tb, 0, stream>>>(Wkv,  wT_kv,  1024, 2048);
    transpose_cvt_kernel<<<dim3(32, 32),  tb, 0, stream>>>(Wq,   wT_q,   1024, 1024);
    transpose_cvt_kernel<<<dim3(32, 32),  tb, 0, stream>>>(Wco,  wT_co,  1024, 1024);
    transpose_cvt_kernel<<<dim3(96, 32),  tb, 0, stream>>>(Wqkv, wT_qkv, 1024, 3072);
    transpose_cvt_kernel<<<dim3(32, 32),  tb, 0, stream>>>(Wso,  wT_so,  1024, 1024);
    transpose_cvt_kernel<<<dim3(128, 32), tb, 0, stream>>>(W1,   wT_w1,  1024, 4096);
    transpose_cvt_kernel<<<dim3(32, 128), tb, 0, stream>>>(W2,   wT_w2,  4096, 1024);

    // ---- cross attention ----
    ln_kernel<<<4096, 256, 0, stream>>>(x,    ln1s, ln1b, lnB);
    ln_kernel<<<4096, 256, 0, stream>>>(keyv, ln2s, ln2b, a2B);
    gemm_bf16_kernel<1, false, false><<<dim3(16, 32), 256, 0, stream>>>(a2B, wT_kv, nullptr, nullptr, kvB, 2048, 1024);
    gemm_bf16_kernel<1, false, false><<<dim3(8, 32),  256, 0, stream>>>(lnB, wT_q,  nullptr, nullptr, qB,  1024, 1024);
    flash_attn_kernel<<<dim3(8, 64), 256, 0, stream>>>(qB, kvB, kvB + 1024, lnB, 1024, 2048);
    gemm_bf16_kernel<0, true, true><<<dim3(8, 32),  256, 0, stream>>>(lnB, wT_co, bco, x, xrB, 1024, 1024);

    // ---- self attention ----
    ln_kernel<<<4096, 256, 0, stream>>>(xrB, ln3s, ln3b, lnB);
    gemm_bf16_kernel<1, false, false><<<dim3(24, 32), 256, 0, stream>>>(lnB, wT_qkv, nullptr, nullptr, kvB, 3072, 1024);
    flash_attn_kernel<<<dim3(8, 64), 256, 0, stream>>>(kvB, kvB + 1024, kvB + 2048, lnB, 3072, 3072);
    gemm_bf16_kernel<0, true, true><<<dim3(8, 32),  256, 0, stream>>>(lnB, wT_so, bso, xrB, x2B, 1024, 1024);

    // ---- MLP (no residual) ----
    ln_kernel<<<4096, 256, 0, stream>>>(x2B, ln4s, ln4b, lnB);
    gemm_bf16_kernel<2, true, false><<<dim3(32, 32), 256, 0, stream>>>(lnB, wT_w1, b1, nullptr, hB, 4096, 1024);
    gemm_bf16_kernel<0, true, false><<<dim3(8, 32),  256, 0, stream>>>(hB,  wT_w2, b2, nullptr, out, 1024, 4096);
}
